// MultiHeadAttention_11252814315825
// MI455X (gfx1250) — matrix operand order repaired and checked by reading, not compiled, not run on hardware
//
#include <hip/hip_runtime.h>
#include <hip/hip_bf16.h>
#include <stdint.h>

// CDNA5 wave32 WMMA fragment types
typedef __attribute__((ext_vector_type(16))) __bf16 v16bf;
typedef __attribute__((ext_vector_type(8)))  float  v8f;

union FragA { v16bf v; __bf16 e[16]; };   // A/B fragment (16 bf16 = 8 VGPRs)
union FragC { v8f  v; float  e[8];  };    // C/D fragment (8 f32  = 8 VGPRs)

#define H_DIM  1024
#define S_LEN  1024
#define BATCH  4
#define NHEAD  16
#define DHEAD  64
#define NEG_MASK (-1.0e12f)

// ---------------------------------------------------------------------------
// GEMM + bias: Y = X @ W + bias
//   X: (M x K) f32 row-major,  W: (K x N) f32 row-major,  bias: (N) f32
//   MODE 0: bf16 output scattered to head-split layout (B, NH, S, DH)
//   MODE 1: f32 output, plain row-major (M x N)
// Block: 256 threads (8 waves). Tile 128x128, K-step 64 (2 WMMA K-subchunks).
// Wave tile 32x64 -> 2(M) x 4(N) accumulators, 16 WMMAs per K-step.
// ---------------------------------------------------------------------------
template<int MODE>
__global__ __launch_bounds__(256)
void gemm_bias_kernel(const float* __restrict__ X,
                      const float* __restrict__ W,
                      const float* __restrict__ bias,
                      void* __restrict__ out,
                      int M, int K, int N)
{
    constexpr int BM = 128, BN = 128, BK = 64;
    __shared__ __bf16 Xs [BM][BK + 8];   // (m, k) row-major
    __shared__ __bf16 Wst[BN][BK + 8];   // (n, k) transposed -> contiguous B-frag reads

    const int tid  = threadIdx.x;
    const int lane = tid & 31;
    const int wid  = tid >> 5;
    const int half = lane >> 4;          // 0: lanes 0-15, 1: lanes 16-31
    const int l16  = lane & 15;
    const int m0 = blockIdx.y * BM;
    const int n0 = blockIdx.x * BN;
    const int waveM = wid & 3;           // 4 waves along M (32 rows each)
    const int waveN = wid >> 2;          // 2 waves along N (64 cols each)

    FragC acc[2][4];
#pragma unroll
    for (int mt = 0; mt < 2; ++mt)
#pragma unroll
        for (int nt = 0; nt < 4; ++nt)
#pragma unroll
            for (int v = 0; v < 8; ++v) acc[mt][nt].e[v] = 0.0f;

    for (int k0 = 0; k0 < K; k0 += BK) {
        __syncthreads();
        // ---- X tile 128x64: f32 global (float4, coalesced) -> bf16 LDS ----
        {
            const float* Xg = X + (size_t)m0 * K + k0;
#pragma unroll
            for (int i = 0; i < 8; ++i) {
                int slot = tid + i * 256;             // 0..2047 = 128 rows * 16 float4
                int r  = slot >> 4;
                int c4 = (slot & 15) * 4;
                const float4 u = *reinterpret_cast<const float4*>(Xg + (size_t)r * K + c4);
                Xs[r][c4 + 0] = (__bf16)u.x;
                Xs[r][c4 + 1] = (__bf16)u.y;
                Xs[r][c4 + 2] = (__bf16)u.z;
                Xs[r][c4 + 3] = (__bf16)u.w;
                if (k0 + BK < K)
                    __builtin_prefetch(Xg + (size_t)r * K + c4 + BK, 0, 1);
            }
        }
        // ---- W tile 64x128: f32 global -> bf16 LDS, transposed (n, k) ----
        {
            const float* Wg = W + (size_t)k0 * N + n0;
#pragma unroll
            for (int i = 0; i < 8; ++i) {
                int slot = tid + i * 256;             // 0..2047 = 64 rows * 32 float4
                int r  = slot >> 5;                   // k
                int c4 = (slot & 31) * 4;             // n
                const float4 u = *reinterpret_cast<const float4*>(Wg + (size_t)r * N + c4);
                Wst[c4 + 0][r] = (__bf16)u.x;
                Wst[c4 + 1][r] = (__bf16)u.y;
                Wst[c4 + 2][r] = (__bf16)u.z;
                Wst[c4 + 3][r] = (__bf16)u.w;
            }
        }
        __syncthreads();

#pragma unroll
        for (int kc = 0; kc < 2; ++kc) {              // two 32-deep WMMA K-subchunks
            // A fragments: 16x32 bf16, ISA layout (K groups of 8 by lane-half)
            FragA afr[2];
#pragma unroll
            for (int mt = 0; mt < 2; ++mt) {
                const __bf16* src = &Xs[waveM * 32 + mt * 16 + l16][kc * 32];
                const int kb = half * 8;
#pragma unroll
                for (int i = 0; i < 8; ++i) afr[mt].e[i]     = src[kb + i];
#pragma unroll
                for (int i = 0; i < 8; ++i) afr[mt].e[8 + i] = src[16 + kb + i];
            }
            // B fragments (same K groups of 8 by lane-half as A) + WMMA
#pragma unroll
            for (int nt = 0; nt < 4; ++nt) {
                FragA bfr;
                const __bf16* src = &Wst[waveN * 64 + nt * 16 + l16][kc * 32];
#pragma unroll
                for (int i = 0; i < 8; ++i) bfr.e[i] = src[half * 8 + i];
#pragma unroll
                for (int i = 0; i < 8; ++i) bfr.e[8 + i] = src[16 + half * 8 + i];
#pragma unroll
                for (int mt = 0; mt < 2; ++mt) {
                    acc[mt][nt].v = __builtin_amdgcn_wmma_f32_16x16x32_bf16(
                        false, afr[mt].v, false, bfr.v, (short)0, acc[mt][nt].v, false, false);
                }
            }
        }
    }

    // ---- epilogue: + bias, scatter ----
#pragma unroll
    for (int nt = 0; nt < 4; ++nt) {
        const int ncol = n0 + waveN * 64 + nt * 16 + l16;
        const float bb = bias[ncol];
#pragma unroll
        for (int mt = 0; mt < 2; ++mt) {
            const int mbase = m0 + waveM * 32 + mt * 16 + half * 8;
#pragma unroll
            for (int v = 0; v < 8; ++v) {
                const int mrow = mbase + v;
                const float val = acc[mt][nt].e[v] + bb;
                if (MODE == 0) {
                    const int bidx = mrow >> 10;            // / S_LEN
                    const int s    = mrow & (S_LEN - 1);
                    const int h    = ncol >> 6;             // / DHEAD
                    const int d    = ncol & (DHEAD - 1);
                    reinterpret_cast<__bf16*>(out)
                        [(((size_t)bidx * NHEAD + h) * S_LEN + s) * DHEAD + d] = (__bf16)val;
                } else {
                    reinterpret_cast<float*>(out)[(size_t)mrow * N + ncol] = val;
                }
            }
        }
    }
}

// ---------------------------------------------------------------------------
// Flash attention (online softmax), one head-block per workgroup.
//   Grid: (S/128, NH, B). 256 threads = 8 waves, each wave owns 16 queries.
//   64-key chunks: 8 WMMA for Q.K^T (4 key tiles x DH=2x32), one shuffle
//   reduction round per chunk (xor 1/2/4/8 within 16-lane halves matches the
//   C-layout row placement), P staged through LDS (C-layout -> A-layout),
//   8 WMMA for P.V. 16 chunk iterations over S=1024.
//   Mask is the reference's additive per-query-row constant (applied pre-max,
//   so masked rows collapse to uniform softmax exactly as in f32 reference).
// ---------------------------------------------------------------------------
__global__ __launch_bounds__(256)
void flash_attention_kernel(const __bf16* __restrict__ Qb,
                            const __bf16* __restrict__ Kb,
                            const __bf16* __restrict__ Vb,
                            const int*    __restrict__ mask,
                            float*        __restrict__ ctx)
{
    constexpr int BK = 64;                    // keys per chunk
    __shared__ __bf16 Qs [128][DHEAD + 8];    // (q, d)
    __shared__ __bf16 Ks [BK][DHEAD + 8];     // (k, d)  contiguous QK^T B-frag reads
    __shared__ __bf16 Vst[DHEAD][BK + 8];     // (d, k)  contiguous PV  B-frag reads
    __shared__ __bf16 Ps [8][16][BK + 8];     // per-wave P staging (q, k)

    const int tid  = threadIdx.x;
    const int lane = tid & 31;
    const int wid  = tid >> 5;
    const int half = lane >> 4;
    const int l16  = lane & 15;
    const int qblk = blockIdx.x, h = blockIdx.y, b = blockIdx.z;
    const int qbase = qblk * 128;

    const size_t headRow = ((size_t)b * NHEAD + h) * S_LEN;
    const __bf16* Qg = Qb + (headRow + qbase) * DHEAD;
    const __bf16* Kg = Kb + headRow * DHEAD;
    const __bf16* Vg = Vb + headRow * DHEAD;

    // ---- cooperative Q tile load: 128 x 64 bf16 (uint2 = 4 elems) ----
#pragma unroll
    for (int i = 0; i < 8; ++i) {
        int slot = tid + i * 256;                 // 0..2047
        int r  = slot >> 4;
        int c4 = (slot & 15) * 4;
        *reinterpret_cast<uint2*>(&Qs[r][c4]) =
            *reinterpret_cast<const uint2*>(Qg + (size_t)r * DHEAD + c4);
    }
    __syncthreads();

    // ---- per-wave Q fragments: two K-chunks over DH=64 ----
    FragA qfr[2];
    {
        const __bf16* src = &Qs[wid * 16 + l16][0];
        const int kb = half * 8;
#pragma unroll
        for (int dc = 0; dc < 2; ++dc) {
#pragma unroll
            for (int i = 0; i < 8; ++i) qfr[dc].e[i]     = src[dc * 32 + kb + i];
#pragma unroll
            for (int i = 0; i < 8; ++i) qfr[dc].e[8 + i] = src[dc * 32 + 16 + kb + i];
        }
    }

    // Row stats: thread holds rows (half*8 + v), v = 0..7 of its wave's tiles.
    float m_i[8], l_i[8], madd[8];
#pragma unroll
    for (int v = 0; v < 8; ++v) { m_i[v] = -3.0e38f; l_i[v] = 0.0f; }
#pragma unroll
    for (int v = 0; v < 8; ++v) {
        const int q = qbase + wid * 16 + half * 8 + v;
        madd[v] = mask[(size_t)b * S_LEN + q] ? 0.0f : NEG_MASK;
    }

    FragC acc[4];                                 // ctx accumulator 16 x 64
#pragma unroll
    for (int nt = 0; nt < 4; ++nt)
#pragma unroll
        for (int v = 0; v < 8; ++v) acc[nt].e[v] = 0.0f;

    for (int kb0 = 0; kb0 < S_LEN; kb0 += BK) {
        __syncthreads();
        // ---- K chunk (64 x 64) -> Ks row-major; V chunk -> Vst transposed ----
#pragma unroll
        for (int i = 0; i < 4; ++i) {
            int slot = tid + i * 256;             // 0..1023
            int r  = slot >> 4;                   // key 0..63
            int c4 = (slot & 15) * 4;             // d
            *reinterpret_cast<uint2*>(&Ks[r][c4]) =
                *reinterpret_cast<const uint2*>(Kg + (size_t)(kb0 + r) * DHEAD + c4);
            const __bf16* vs = Vg + (size_t)(kb0 + r) * DHEAD + c4;
            Vst[c4 + 0][r] = vs[0];
            Vst[c4 + 1][r] = vs[1];
            Vst[c4 + 2][r] = vs[2];
            Vst[c4 + 3][r] = vs[3];
        }
        __syncthreads();

        // ---- scores S = Q . K^T  (16 queries x 64 keys) ----
        FragC sc[4];
#pragma unroll
        for (int kt = 0; kt < 4; ++kt)
#pragma unroll
            for (int v = 0; v < 8; ++v) sc[kt].e[v] = 0.0f;
#pragma unroll
        for (int kt = 0; kt < 4; ++kt) {
            const __bf16* src = &Ks[kt * 16 + l16][0];
#pragma unroll
            for (int dc = 0; dc < 2; ++dc) {
                FragA bfr;
#pragma unroll
                for (int i = 0; i < 8; ++i) bfr.e[i] = src[dc * 32 + half * 8 + i];
#pragma unroll
                for (int i = 0; i < 8; ++i) bfr.e[8 + i] = src[dc * 32 + 16 + half * 8 + i];
                sc[kt].v = __builtin_amdgcn_wmma_f32_16x16x32_bf16(
                    false, qfr[dc].v, false, bfr.v, (short)0, sc[kt].v, false, false);
            }
        }

        // ---- online softmax update (scale 1/sqrt(64) = 0.125, additive mask) ----
        float scale[8];
#pragma unroll
        for (int v = 0; v < 8; ++v) {
            float s[4];
#pragma unroll
            for (int kt = 0; kt < 4; ++kt) s[kt] = sc[kt].e[v] * 0.125f + madd[v];
            float tm = fmaxf(fmaxf(s[0], s[1]), fmaxf(s[2], s[3]));
            tm = fmaxf(tm, __shfl_xor(tm, 1, 32));
            tm = fmaxf(tm, __shfl_xor(tm, 2, 32));
            tm = fmaxf(tm, __shfl_xor(tm, 4, 32));
            tm = fmaxf(tm, __shfl_xor(tm, 8, 32));
            const float nm = fmaxf(m_i[v], tm);
            float ts = 0.0f;
#pragma unroll
            for (int kt = 0; kt < 4; ++kt) {
                const float p = __expf(s[kt] - nm);
                sc[kt].e[v] = p;
                ts += p;
            }
            ts += __shfl_xor(ts, 1, 32);
            ts += __shfl_xor(ts, 2, 32);
            ts += __shfl_xor(ts, 4, 32);
            ts += __shfl_xor(ts, 8, 32);
            scale[v] = __expf(m_i[v] - nm);
            l_i[v] = l_i[v] * scale[v] + ts;
            m_i[v] = nm;
        }
#pragma unroll
        for (int nt = 0; nt < 4; ++nt)
#pragma unroll
            for (int v = 0; v < 8; ++v) acc[nt].e[v] *= scale[v];

        // ---- stage P (16x64, C-layout) to LDS, reload in A-fragment layout ----
#pragma unroll
        for (int kt = 0; kt < 4; ++kt)
#pragma unroll
            for (int v = 0; v < 8; ++v)
                Ps[wid][half * 8 + v][kt * 16 + l16] = (__bf16)sc[kt].e[v];
        __syncthreads();
        FragA pfr[2];                             // two 32-deep key subchunks
        {
            const __bf16* src = &Ps[wid][l16][0];
            const int kb = half * 8;
#pragma unroll
            for (int pc = 0; pc < 2; ++pc) {
#pragma unroll
                for (int i = 0; i < 8; ++i) pfr[pc].e[i]     = src[pc * 32 + kb + i];
#pragma unroll
                for (int i = 0; i < 8; ++i) pfr[pc].e[8 + i] = src[pc * 32 + 16 + kb + i];
            }
        }

        // ---- ctx += P . V ----
#pragma unroll
        for (int nt = 0; nt < 4; ++nt) {
            const __bf16* src = &Vst[nt * 16 + l16][0];
#pragma unroll
            for (int pc = 0; pc < 2; ++pc) {
                FragA bfr;
#pragma unroll
                for (int i = 0; i < 8; ++i) bfr.e[i] = src[pc * 32 + half * 8 + i];
#pragma unroll
                for (int i = 0; i < 8; ++i) bfr.e[8 + i] = src[pc * 32 + 16 + half * 8 + i];
                acc[nt].v = __builtin_amdgcn_wmma_f32_16x16x32_bf16(
                    false, pfr[pc].v, false, bfr.v, (short)0, acc[nt].v, false, false);
            }
        }
    }

    // ---- normalize and store ctx in (B, S, NH, DH) f32 ----
#pragma unroll
    for (int v = 0; v < 8; ++v) {
        const float inv = 1.0f / l_i[v];
        const int q = qbase + wid * 16 + half * 8 + v;
#pragma unroll
        for (int nt = 0; nt < 4; ++nt) {
            const int d = nt * 16 + l16;
            ctx[(((size_t)b * S_LEN + q) * NHEAD + h) * DHEAD + d] = acc[nt].e[v] * inv;
        }
    }
}

// ---------------------------------------------------------------------------
extern "C" void kernel_launch(void* const* d_in, const int* in_sizes, int n_in,
                              void* d_out, int out_size, void* d_ws, size_t ws_size,
                              hipStream_t stream)
{
    const float* query = (const float*)d_in[0];
    const float* key_t = (const float*)d_in[1];
    const float* value = (const float*)d_in[2];
    const int*   mask  = (const int*)  d_in[3];
    const float* Wq = (const float*)d_in[4];  const float* bq = (const float*)d_in[5];
    const float* Wk = (const float*)d_in[6];  const float* bk = (const float*)d_in[7];
    const float* Wv = (const float*)d_in[8];  const float* bv = (const float*)d_in[9];
    const float* Wo = (const float*)d_in[10]; const float* bo = (const float*)d_in[11];

    // Workspace layout: Q,K,V as bf16 head-split (8 MB each), ctx f32 (16 MB).
    char* ws = (char*)d_ws;
    const size_t qkvElems = (size_t)BATCH * NHEAD * S_LEN * DHEAD;   // 4 Mi
    __bf16* Qb  = (__bf16*)(ws);
    __bf16* Kb  = (__bf16*)(ws + qkvElems * 2);
    __bf16* Vb  = (__bf16*)(ws + qkvElems * 4);
    float*  ctx = (float*) (ws + qkvElems * 6);

    const int M = BATCH * S_LEN;               // 4096
    dim3 gemmGrid(H_DIM / 128, M / 128);       // (8, 32)
    dim3 attnGrid(S_LEN / 128, NHEAD, BATCH);  // (8, 16, 4)

    gemm_bias_kernel<0><<<gemmGrid, 256, 0, stream>>>(query, Wq, bq, Qb, M, H_DIM, H_DIM);
    gemm_bias_kernel<0><<<gemmGrid, 256, 0, stream>>>(key_t, Wk, bk, Kb, M, H_DIM, H_DIM);
    gemm_bias_kernel<0><<<gemmGrid, 256, 0, stream>>>(value, Wv, bv, Vb, M, H_DIM, H_DIM);

    flash_attention_kernel<<<attnGrid, 256, 0, stream>>>(Qb, Kb, Vb, mask, ctx);

    gemm_bias_kernel<1><<<gemmGrid, 256, 0, stream>>>(ctx, Wo, bo, d_out, M, H_DIM, H_DIM);
}
